// LearnableNeuralLogicLayer_39290360823856
// MI455X (gfx1250) — matrix operand order repaired and checked by reading, not compiled, not run on hardware
//
#include <hip/hip_runtime.h>
#include <hip/hip_bf16.h>
#include <math.h>

typedef __attribute__((ext_vector_type(16))) _Float16 v16h;
typedef __attribute__((ext_vector_type(8)))  float    v8f;
typedef __attribute__((ext_vector_type(8)))  float    float8;

#define NFEAT   4096
#define NCL     40
#define CPAD    48
#define NKT     (NFEAT / 32)   // 128 k-tiles of 32
#define NCT     (CPAD / 16)    // 3 column tiles
#define KSPLIT  4              // K split: 4 partial GEMMs of K=1024 (32 k-tiles)
#define KTPS    (NKT / KSPLIT) // 32 k-tiles per K-slice
#define EPSF    1e-8f
#define ATHRESH 1e-6f

// ---------------------------------------------------------------------------
// Kernel 1: build ternary weight matrix in WMMA-B-fragment swizzled layout,
// and per-column counts. One block per (padded) clause column.
// Swizzle: (K=f, N=c) -> bsw[(((f>>5)*3 + (c>>4))*32 + ((f>>4)&1)*16 + (c&15))*16 + (f&15)]
// ---------------------------------------------------------------------------
__global__ __launch_bounds__(128) void logic_prep_kernel(
    const float* __restrict__ cw,   // (F, 40) clause_weights
    const float* __restrict__ ml,   // (F, 40) mask_logits
    const float* __restrict__ u,    // (F, 40) uniform noise
    const float* __restrict__ temp, // scalar
    _Float16*    __restrict__ bsw,  // swizzled ternary B, 128*3*512 halfs
    float*       __restrict__ cnt,  // (48,)
    float*       __restrict__ cneg) // (48,)
{
    const int c   = blockIdx.x;      // 0..47 (40..47 are zero padding)
    const int tid = threadIdx.x;     // 0..127
    const float invT = 1.0f / (temp[0] + EPSF);

    const int ct = c >> 4;
    const int n  = c & 15;

    float myCnt = 0.0f, myNeg = 0.0f;

    for (int f = tid; f < NFEAT; f += 128) {
        float tern = 0.0f;
        if (c < NCL) {
            const int idx = f * NCL + c;
            const float uu = u[idx];
            const float logit = __logf(uu + EPSF) - __logf(1.0f - uu + EPSF) + ml[idx];
            const float s = 1.0f / (1.0f + __expf(-logit * invT));
            const float w = tanhf(cw[idx] * invT) * s;
            const bool active = fabsf(w) > ATHRESH;
            const bool pos = active && (w > 0.0f);
            const bool neg = active && !(w > 0.0f);
            tern   = pos ? 1.0f : (neg ? -1.0f : 0.0f);
            myCnt += active ? 1.0f : 0.0f;
            myNeg += neg    ? 1.0f : 0.0f;
        }
        const int kt   = f >> 5;
        const int half = (f >> 4) & 1;
        const int e    = f & 15;
        const size_t off = (((size_t)(kt * NCT + ct)) * 32 + half * 16 + n) * 16 + e;
        bsw[off] = (_Float16)tern;
    }

    __shared__ float s1[128];
    __shared__ float s2[128];
    s1[tid] = myCnt;
    s2[tid] = myNeg;
    __syncthreads();
    for (int st = 64; st > 0; st >>= 1) {
        if (tid < st) { s1[tid] += s1[tid + st]; s2[tid] += s2[tid + st]; }
        __syncthreads();
    }
    if (tid == 0) { cnt[c] = s1[0]; cneg[c] = s2[0]; }
}

// ---------------------------------------------------------------------------
// Kernel 2: partial WMMA GEMM over one K-slice:  part[ks] = x[:, ksK] @ tern[ksK, :]
// 256 threads = 8 waves/block; each wave: 32 rows x 48 cols (2x3 tiles), K=1024.
// f32 features split into hi/lo f16 -> two v_wmma_f32_16x16x32_f16 per tile.
// Feature loads are non-temporal (pure stream, no reuse) to keep L2 for B+partials.
// ---------------------------------------------------------------------------
__global__ __launch_bounds__(256) void logic_gemm_kernel(
    const float*    __restrict__ feat,  // (B, 4096)
    const _Float16* __restrict__ bsw,   // swizzled ternary B
    float*          __restrict__ part,  // (KSPLIT, B, 48) partial sums
    int batch)
{
    const int lane  = threadIdx.x & 31;
    const int wave  = threadIdx.x >> 5;
    const int tl    = lane & 15;
    const int half  = lane >> 4;
    const int gwave = blockIdx.x * 8 + wave;
    const int rowBase = gwave * 32;
    const int ks    = blockIdx.y;           // K-slice 0..KSPLIT-1
    const int ktLo  = ks * KTPS;

    const v16h* __restrict__ bswv = (const v16h*)bsw;

    v8f acc[2][NCT];
#pragma unroll
    for (int mt = 0; mt < 2; ++mt)
#pragma unroll
        for (int ct = 0; ct < NCT; ++ct)
            acc[mt][ct] = (v8f)(0.0f);

    for (int kk = 0; kk < KTPS; ++kk) {
        const int kt = ktLo + kk;
        const int K0 = kt * 32;

        // B fragments for the 3 column tiles (already swizzled: 1 v16h/lane)
        v16h bf[NCT];
#pragma unroll
        for (int ct = 0; ct < NCT; ++ct)
            bf[ct] = bswv[(size_t)(kt * NCT + ct) * 32 + lane];

#pragma unroll
        for (int mt = 0; mt < 2; ++mt) {
            const int m = rowBase + mt * 16 + tl;
            const float* rp = feat + (size_t)m * NFEAT + K0 + half * 16;
            const float8 x0 = __builtin_nontemporal_load((const float8*)(rp));      // K0+16h..+7
            const float8 x1 = __builtin_nontemporal_load((const float8*)(rp + 8));  // K0+16h+8..+15

            v16h ahi, alo;
#pragma unroll
            for (int e = 0; e < 8; ++e) {
                float x = x0[e];
                _Float16 h = (_Float16)x;
                ahi[e] = h;
                alo[e] = (_Float16)(x - (float)h);
                x = x1[e];
                h = (_Float16)x;
                ahi[8 + e] = h;
                alo[8 + e] = (_Float16)(x - (float)h);
            }

#pragma unroll
            for (int ct = 0; ct < NCT; ++ct) {
                acc[mt][ct] = __builtin_amdgcn_wmma_f32_16x16x32_f16(
                    false, ahi, false, bf[ct], (short)0, acc[mt][ct], false, false);
                acc[mt][ct] = __builtin_amdgcn_wmma_f32_16x16x32_f16(
                    false, alo, false, bf[ct], (short)0, acc[mt][ct], false, false);
            }
        }
    }

    // Store raw partial accumulators: part[ks][row][c]. D[r]: row=8*half+r, col=tl.
    float* pbase = part + (size_t)ks * batch * CPAD;
#pragma unroll
    for (int mt = 0; mt < 2; ++mt) {
#pragma unroll
        for (int ct = 0; ct < NCT; ++ct) {
            const int c = ct * 16 + tl;
#pragma unroll
            for (int r = 0; r < 8; ++r) {
                const int row = rowBase + mt * 16 + half * 8 + r;
                pbase[(size_t)row * CPAD + c] = acc[mt][ct][r];
            }
        }
    }
}

// ---------------------------------------------------------------------------
// Kernel 3: reduce K-slices, apply cnt_neg / 1/cnt, sum 10 clauses per action.
// One thread per batch row. Partials are L2-resident (12.6 MB << 192 MB).
// ---------------------------------------------------------------------------
__global__ __launch_bounds__(256) void logic_finish_kernel(
    const float* __restrict__ part,  // (KSPLIT, B, 48)
    const float* __restrict__ cnt,   // (48,)
    const float* __restrict__ cneg,  // (48,)
    float*       __restrict__ out,   // (B, 4)
    int batch)
{
    const int r = blockIdx.x * 256 + threadIdx.x;
    if (r >= batch) return;

    float as0 = 0.0f, as1 = 0.0f, as2 = 0.0f, as3 = 0.0f;
#pragma unroll
    for (int c = 0; c < NCL; ++c) {
        float num = cneg[c];
#pragma unroll
        for (int ksl = 0; ksl < KSPLIT; ++ksl)
            num += part[((size_t)ksl * batch + r) * CPAD + c];
        const float cc = cnt[c];
        const float sat = (cc > 0.0f) ? num / fmaxf(cc, 1.0f) : 1.0f;
        if      (c < 10) as0 += sat;
        else if (c < 20) as1 += sat;
        else if (c < 30) as2 += sat;
        else             as3 += sat;
    }
    float* o = out + (size_t)r * 4;
    o[0] = as0; o[1] = as1; o[2] = as2; o[3] = as3;
}

// ---------------------------------------------------------------------------
// Launcher
// ---------------------------------------------------------------------------
extern "C" void kernel_launch(void* const* d_in, const int* in_sizes, int n_in,
                              void* d_out, int out_size, void* d_ws, size_t ws_size,
                              hipStream_t stream) {
    const float* feat = (const float*)d_in[0];   // (B, 4096)
    const float* cw   = (const float*)d_in[1];   // (4096, 40)
    const float* ml   = (const float*)d_in[2];   // (4096, 40)
    const float* u    = (const float*)d_in[3];   // (4096, 40)
    const float* temp = (const float*)d_in[4];   // scalar

    const int batch = in_sizes[0] / NFEAT;       // 16384

    const size_t BSW_BYTES = (size_t)NKT * NCT * 32 * 16 * sizeof(_Float16); // 393216
    _Float16* bsw  = (_Float16*)d_ws;
    float*    cnt  = (float*)((char*)d_ws + BSW_BYTES);
    float*    cneg = (float*)((char*)d_ws + BSW_BYTES + 256);
    float*    part = (float*)((char*)d_ws + ((BSW_BYTES + 65535) & ~(size_t)65535));
    // part at 64KB-aligned offset (384 KB); size KSPLIT*batch*48*4 = 12.6 MB

    logic_prep_kernel<<<CPAD, 128, 0, stream>>>(cw, ml, u, temp, bsw, cnt, cneg);

    dim3 ggrid(batch / 256, KSPLIT);             // 64 x 4 blocks, 256 thr = 8 waves
    logic_gemm_kernel<<<ggrid, 256, 0, stream>>>(feat, bsw, part, batch);

    logic_finish_kernel<<<(batch + 255) / 256, 256, 0, stream>>>(part, cnt, cneg,
                                                                 (float*)d_out, batch);
}
